// ConvLSTMBlock_70720931496817
// MI455X (gfx1250) — compile-verified, matrix operand order checked by reading, not run on hardware
//
#include <hip/hip_runtime.h>
#include <hip/hip_bf16.h>
#include <cstdint>
#include <cstddef>

// ---------------------------------------------------------------------------
// ConvLSTM2D + BN + MaxPool2x2 for MI455X (gfx1250, wave32, WMMA).
// Compute-bound (~72 GFLOP vs ~60 MB HBM traffic) => fast XDL path:
// v_wmma_f32_16x16x32_f16 (f16 inputs, f32 accumulation), B-fragments staged
// in LDS (320 KB/WGP) instead of re-fetching from L2 per WMMA.
// ---------------------------------------------------------------------------

typedef _Float16 half_t;
typedef __attribute__((ext_vector_type(16))) _Float16 v16h;
typedef __attribute__((ext_vector_type(8)))  _Float16 v8h;
typedef __attribute__((ext_vector_type(8)))  float    v8f;

#define BATCH   8
#define TSTEPS  20
#define HH      64
#define WW      64
#define CIN     16
#define FF      32
#define OC      128          // 4 gates * FF

#define XF16_ELEMS   ((size_t)BATCH*TSTEPS*HH*WW*CIN)   // 10,485,760 f16
#define HBUF_ELEMS   ((size_t)BATCH*HH*WW*FF)           //  1,048,576 f16
#define CBUF_ELEMS   ((size_t)BATCH*HH*WW*FF)           //  f32
#define WXP_ELEMS    (5*8*32*16)                        // 20,480 f16
#define WHP_ELEMS    (9*8*32*16)                        // 36,864 f16
#define WALL_ELEMS   (WXP_ELEMS + WHP_ELEMS)            // 57,344 f16 = 112 KB

// ---------------------------------------------------------------------------
__global__ void cvt_x_kernel(const float* __restrict__ x, half_t* __restrict__ xf,
                             size_t n) {
    size_t i = (size_t)blockIdx.x * blockDim.x + threadIdx.x;
    size_t stride = (size_t)gridDim.x * blockDim.x;
    for (; i < n; i += stride) xf[i] = (half_t)x[i];
}

__global__ void zero_u32_kernel(uint32_t* __restrict__ p, size_t n) {
    size_t i = (size_t)blockIdx.x * blockDim.x + threadIdx.x;
    size_t stride = (size_t)gridDim.x * blockDim.x;
    for (; i < n; i += stride) p[i] = 0u;
}

// ---------------------------------------------------------------------------
// Pack conv weights into the per-lane B-matrix layout of
// V_WMMA_F32_16X16X32_F16 (32x16, K x N), f16.
// Lane L holds column N = L&15; half-element e of the v16h maps to
//   K = e + (e>=8 ? 8 : 0) + (L>=16 ? 8 : 0)
// Input-conv K-chunks pair two 3x3 taps (16 ch each); tap 9 is zero padding.
// Recurrent-conv chunks are one tap each (32 ch).
// ---------------------------------------------------------------------------
__global__ void pack_weights_kernel(const float* __restrict__ kern,   // [3,3,16,128]
                                    const float* __restrict__ rkern,  // [3,3,32,128]
                                    half_t* __restrict__ wxp,
                                    half_t* __restrict__ whp) {
    int i = blockIdx.x * blockDim.x + threadIdx.x;
    if (i < (int)WXP_ELEMS) {
        int e    = i & 15;
        int lane = (i >> 4) & 31;
        int nt   = (i >> 9) & 7;
        int P    = i >> 12;
        int k    = e + (e >= 8 ? 8 : 0) + (lane >> 4) * 8;   // 0..31
        int tap  = P * 2 + (k >> 4);
        half_t v = (half_t)0.0f;
        if (tap < 9) {
            int cin = k & 15;
            int oc  = nt * 16 + (lane & 15);
            v = (half_t)kern[((size_t)tap * CIN + cin) * OC + oc];
        }
        wxp[i] = v;
    }
    int j = i - (int)WXP_ELEMS;
    if (j >= 0 && j < (int)WHP_ELEMS) {
        int e    = j & 15;
        int lane = (j >> 4) & 31;
        int nt   = (j >> 9) & 7;
        int tap  = j >> 12;
        int k    = e + (e >= 8 ? 8 : 0) + (lane >> 4) * 8;   // recurrent channel
        int oc   = nt * 16 + (lane & 15);
        whp[j]   = (half_t)rkern[((size_t)tap * FF + k) * OC + oc];
    }
}

// ---------------------------------------------------------------------------
// One ConvLSTM time step, fused with BN + 2x2 max pool.
// Grid: 256 workgroups = B(8) * rows/4(16) * cols/32(2); block = 256 (8 waves).
// Wave w handles row r0+(w>>1), cols c0+16*(w&1) .. +15 (16 pixels, M-dim),
// all 128 out-channels (8 WMMA N-tiles). Packed weights live in LDS (112 KB,
// staged once per workgroup, amortized over 8 waves; 2 WGs/WGP fit in 320 KB).
// The pooling tile reuses the weight LDS after the GEMM phase.
// ---------------------------------------------------------------------------
struct __align__(32) SMem {
    union {
        uint4  raw[WALL_ELEMS / 8];       // 112 KB staging view
        half_t wts[WALL_ELEMS];           // packed B fragments
        float  pool[4][32][FF];           // 16 KB h tile for pooling (reused)
    };
};

__global__ __launch_bounds__(256)
void convlstm_step_kernel(const half_t* __restrict__ xf16,   // [B,T,H,W,CIN] f16
                          const half_t* __restrict__ hprev,  // [B,H,W,F]    f16
                          half_t*       __restrict__ hnext,  // [B,H,W,F]    f16
                          float*        __restrict__ cstate, // [B,H,W,F]    f32
                          const half_t* __restrict__ wall,   // wxp|whp packed
                          const float*  __restrict__ bias,   // [128]
                          const float*  __restrict__ gamma_, // [32]
                          const float*  __restrict__ beta_,  // [32]
                          const float*  __restrict__ mmean,  // [32]
                          const float*  __restrict__ mvar,   // [32]
                          float*        __restrict__ out,    // [B*T,32,32,F]
                          int t) {
    __shared__ SMem smem;

    const int wg  = blockIdx.x;            // 0..255
    const int b   = wg >> 5;               // / 32
    const int wr4 = (wg >> 1) & 15;        // 4-row stripe index
    const int wc  = wg & 1;                // col half
    const int r0  = wr4 * 4;
    const int c0  = wc * 32;

    const int tid   = threadIdx.x;
    const int wave  = tid >> 5;            // 0..7
    const int lane  = tid & 31;
    const int nIdx  = lane & 15;           // M (A-frag) / N (C-frag) index
    const int hi    = lane >> 4;           // lane-group selects K offset +8
    const int chOff = hi * 8;

    const int r   = r0 + (wave >> 1);
    const int cb  = c0 + (wave & 1) * 16;
    const int col = cb + nIdx;             // this lane's A-matrix pixel column

    // ---- stage packed weights into LDS (one pass, 8 waves cooperate) ----
    {
        const uint4* gsrc = (const uint4*)wall;
        for (int i = tid; i < WALL_ELEMS / 8; i += 256)
            smem.raw[i] = gsrc[i];
    }
    __syncthreads();

    const size_t bt = (size_t)b * TSTEPS + t;
    const half_t* xbase = xf16 + bt * (size_t)HH * WW * CIN;
    const half_t* hbase = hprev + (size_t)b * HH * WW * FF;

    union VU { v16h v; v8h h[2]; };

    // ---- A fragments: input-conv chunks (taps paired, 16 ch each) ----
    v16h Ax[5];
#pragma unroll
    for (int P = 0; P < 5; ++P) {
        VU u;
#pragma unroll
        for (int q = 0; q < 2; ++q) {
            int tap = P * 2 + q;
            v8h val = {};
            if (tap < 9) {
                int rr = r + tap / 3 - 1, cc = col + tap % 3 - 1;
                if (rr >= 0 && rr < HH && cc >= 0 && cc < WW)
                    val = *(const v8h*)(xbase + ((size_t)rr * WW + cc) * CIN + chOff);
            }
            u.h[q] = val;
        }
        Ax[P] = u.v;
    }
    // ---- A fragments: recurrent-conv chunks (one tap, 32 ch) ----
    v16h Ah[9];
#pragma unroll
    for (int tap = 0; tap < 9; ++tap) {
        int rr = r + tap / 3 - 1, cc = col + tap % 3 - 1;
        VU u;
        v8h lo = {}, up = {};
        if (rr >= 0 && rr < HH && cc >= 0 && cc < WW) {
            const half_t* p = hbase + ((size_t)rr * WW + cc) * FF;
            lo = *(const v8h*)(p + chOff);
            up = *(const v8h*)(p + 16 + chOff);
        }
        u.h[0] = lo; u.h[1] = up;
        Ah[tap] = u.v;
    }

    // ---- implicit GEMM: 112 x v_wmma_f32_16x16x32_f16, B from LDS ----
    const half_t* lwx = smem.wts;
    const half_t* lwh = smem.wts + WXP_ELEMS;
    v8f acc[8];
#pragma unroll
    for (int nt = 0; nt < 8; ++nt) {
        v8f a = {};
#pragma unroll
        for (int P = 0; P < 5; ++P) {
            v16h bf = *(const v16h*)(lwx + ((P * 8 + nt) * 32 + lane) * 16);
            a = __builtin_amdgcn_wmma_f32_16x16x32_f16(false, Ax[P], false, bf,
                                                       (short)0, a, false, false);
        }
#pragma unroll
        for (int tp = 0; tp < 9; ++tp) {
            v16h bf = *(const v16h*)(lwh + ((tp * 8 + nt) * 32 + lane) * 16);
            a = __builtin_amdgcn_wmma_f32_16x16x32_f16(false, Ah[tp], false, bf,
                                                       (short)0, a, false, false);
        }
        acc[nt] = a;
    }

    // all waves must be done reading weight LDS before it is reused for pooling
    __syncthreads();

    // ---- gates + cell update (per lane: 8 pixels x 2 channels) ----
    // C/D layout: lane holds oc = nt*16 + (lane&15); VGPR v holds pixel v+hi*8.
    float*  cbase = cstate + (size_t)b * HH * WW * FF;
    half_t* hnb   = hnext  + (size_t)b * HH * WW * FF;

#pragma unroll
    for (int hf = 0; hf < 2; ++hf) {
        const int ch = hf * 16 + nIdx;          // channel within F
        const float bi  = bias[ 0 + ch];
        const float bfg = bias[32 + ch];
        const float bc  = bias[64 + ch];
        const float bo  = bias[96 + ch];
#pragma unroll
        for (int v = 0; v < 8; ++v) {
            const int mm   = v + hi * 8;         // pixel within the 16-tile
            const int ocol = cb + mm;
            const size_t sidx = ((size_t)r * WW + ocol) * FF + ch;
            float zi = acc[0 + hf][v] + bi;
            float zf = acc[2 + hf][v] + bfg;
            float zc = acc[4 + hf][v] + bc;
            float zo = acc[6 + hf][v] + bo;
            float ig = fminf(fmaxf(0.2f * zi + 0.5f, 0.0f), 1.0f);
            float fg = fminf(fmaxf(0.2f * zf + 0.5f, 0.0f), 1.0f);
            float og = fminf(fmaxf(0.2f * zo + 0.5f, 0.0f), 1.0f);
            float cold = cbase[sidx];
            float cnew = fg * cold + ig * tanhf(zc);
            float hout = og * tanhf(cnew);
            cbase[sidx] = cnew;
            hnb[sidx]   = (half_t)hout;
            smem.pool[wave >> 1][(wave & 1) * 16 + mm][ch] = hout;
        }
    }
    __syncthreads();

    // ---- fused BatchNorm (inference) + 2x2 max pool ----
    // 2 pooled rows x 16 pooled cols x 32 ch = 1024 outputs, 256 threads.
#pragma unroll
    for (int q = 0; q < 4; ++q) {
        int o  = q * 256 + tid;
        int ch = o & 31;
        int pc = (o >> 5) & 15;
        int pr = o >> 9;                     // 0..1
        float inv = rsqrtf(mvar[ch] + 1e-3f) * gamma_[ch];
        float sh  = beta_[ch] - mmean[ch] * inv;
        float m0 = smem.pool[2 * pr + 0][2 * pc + 0][ch] * inv + sh;
        float m1 = smem.pool[2 * pr + 0][2 * pc + 1][ch] * inv + sh;
        float m2 = smem.pool[2 * pr + 1][2 * pc + 0][ch] * inv + sh;
        float m3 = smem.pool[2 * pr + 1][2 * pc + 1][ch] * inv + sh;
        float mv = fmaxf(fmaxf(m0, m1), fmaxf(m2, m3));
        out[((bt * 32 + (wr4 * 2 + pr)) * 32 + (wc * 16 + pc)) * FF + ch] = mv;
    }
}

// ---------------------------------------------------------------------------
// Host-side launch. Workspace (~30 MB): x_f16 | h_ping | h_pong | c_f32 | Wx|Wh
// ---------------------------------------------------------------------------
extern "C" void kernel_launch(void* const* d_in, const int* in_sizes, int n_in,
                              void* d_out, int out_size, void* d_ws, size_t ws_size,
                              hipStream_t stream) {
    const float* x     = (const float*)d_in[0];
    const float* kern  = (const float*)d_in[1];
    const float* rkern = (const float*)d_in[2];
    const float* bias  = (const float*)d_in[3];
    const float* gam   = (const float*)d_in[4];
    const float* bet   = (const float*)d_in[5];
    const float* mmean = (const float*)d_in[6];
    const float* mvar  = (const float*)d_in[7];
    float* out = (float*)d_out;

    char* ws = (char*)d_ws;
    const size_t off_x   = 0;
    const size_t off_ha  = off_x  + XF16_ELEMS * sizeof(half_t);   // 20,971,520
    const size_t off_hb  = off_ha + HBUF_ELEMS * sizeof(half_t);   // + 2 MB
    const size_t off_c   = off_hb + HBUF_ELEMS * sizeof(half_t);   // + 2 MB
    const size_t off_wx  = off_c  + CBUF_ELEMS * sizeof(float);    // + 4 MB
    const size_t off_wh  = off_wx + WXP_ELEMS  * sizeof(half_t);

    half_t* xf16 = (half_t*)(ws + off_x);
    half_t* h_a  = (half_t*)(ws + off_ha);
    half_t* h_b  = (half_t*)(ws + off_hb);
    float*  cst  = (float*) (ws + off_c);
    half_t* wxp  = (half_t*)(ws + off_wx);   // wxp|whp contiguous = "wall"
    half_t* whp  = (half_t*)(ws + off_wh);

    // 1) convert x to f16
    cvt_x_kernel<<<2048, 256, 0, stream>>>(x, xf16, XF16_ELEMS);
    // 2) zero h ping-pong + c state (contiguous region: 2*2MB + 4MB)
    zero_u32_kernel<<<2048, 256, 0, stream>>>(
        (uint32_t*)h_a,
        (2 * HBUF_ELEMS * sizeof(half_t) + CBUF_ELEMS * sizeof(float)) / 4);
    // 3) pack weights into WMMA B-fragment layout
    {
        int total = (int)WALL_ELEMS;
        pack_weights_kernel<<<(total + 255) / 256, 256, 0, stream>>>(kern, rkern,
                                                                     wxp, whp);
    }
    // 4) recurrent scan: one step kernel per t (ping-pong h for halo safety)
    for (int t = 0; t < TSTEPS; ++t) {
        const half_t* hp = (t & 1) ? h_b : h_a;
        half_t*       hn = (t & 1) ? h_a : h_b;
        convlstm_step_kernel<<<256, 256, 0, stream>>>(
            xf16, hp, hn, cst, wxp, bias, gam, bet, mmean, mvar, out, t);
    }
}
